// GraphSSM_20899310862683
// MI455X (gfx1250) — matrix operand order repaired and checked by reading, not compiled, not run on hardware
//
#include <hip/hip_runtime.h>
#include <hip/hip_bf16.h>
#include <math.h>

// ---------------------------------------------------------------------------
// Mamba block for MI455X (gfx1250, wave32, WMMA).
// GEMM operands are pre-converted to f16 once (streaming kernels), so the
// WMMA inner loop is pure b128 loads + v_wmma with double-buffered fragments
// (no converts, no hazard NOPs from register reuse). f32 accumulate everywhere.
// ---------------------------------------------------------------------------

typedef __attribute__((ext_vector_type(16))) _Float16 v16h;
typedef __attribute__((ext_vector_type(8)))  _Float16 v8h;
typedef __attribute__((ext_vector_type(4)))  _Float16 v4h;
typedef __attribute__((ext_vector_type(8)))  float    v8f;

#define DM   768
#define DI   1536
#define LSEQ 2048
#define NST  16
#define DTR  48
#define SSMW 80   // DTR + 2*NST
#define KDT  64   // dt_proj K padded to 64

// A fragment (16x32 f16): lane = row M=lane&15, 16 contiguous halves at
// (k0 + 16*(lane>>4)).
__device__ __forceinline__ v16h load_a_frag16(const _Float16* __restrict__ p) {
    return *(const v16h*)(p);
}

// B fragment (32x16 f16): lane = col N=lane&15, 16 contiguous halves at
// (k0 + 16*(lane>>4)).
__device__ __forceinline__ v16h load_b_frag16(const _Float16* __restrict__ p) {
    return *(const v16h*)(p);
}

// EPI: 0 = row-major store; 1 = transposed store; 2 = transposed + softplus(x+bias)
// K must be a multiple of 64 (two-stage ping-pong per iteration).
template<int MT, int NT, int K, int EPI>
__global__ __launch_bounds__(32)
void wmma_gemm(const _Float16* __restrict__ A, int lda,
               const _Float16* __restrict__ Bw, int ldb,
               const float* __restrict__ bias,
               float* __restrict__ Dst, int ldd)
{
    const int lane   = threadIdx.x;            // one wave per block
    const int m0     = blockIdx.y * (16 * MT);
    const int n0     = blockIdx.x * (16 * NT);
    const int khalfA = (lane >> 4) * 16;
    const int khalfB = (lane >> 4) * 16;

    const _Float16* aptr[MT];
#pragma unroll
    for (int mt = 0; mt < MT; ++mt)
        aptr[mt] = A + (size_t)(m0 + mt * 16 + (lane & 15)) * lda + khalfA;
    const _Float16* bptr[NT];
#pragma unroll
    for (int nt = 0; nt < NT; ++nt)
        bptr[nt] = Bw + (size_t)(n0 + nt * 16 + (lane & 15)) * ldb + khalfB;

    v8f acc[MT][NT];
#pragma unroll
    for (int mt = 0; mt < MT; ++mt)
#pragma unroll
        for (int nt = 0; nt < NT; ++nt) acc[mt][nt] = (v8f){};

    v16h a0[MT], a1[MT], b0[NT], b1[NT];
    // prologue: stage 0 fragments at k=0
#pragma unroll
    for (int mt = 0; mt < MT; ++mt) a0[mt] = load_a_frag16(aptr[mt]);
#pragma unroll
    for (int nt = 0; nt < NT; ++nt) b0[nt] = load_b_frag16(bptr[nt]);

    for (int k0 = 0; k0 < K; k0 += 64) {
        // stage 1 loads (k0+32) — write regs disjoint from stage-0 WMMA sources
#pragma unroll
        for (int mt = 0; mt < MT; ++mt) a1[mt] = load_a_frag16(aptr[mt] + k0 + 32);
#pragma unroll
        for (int nt = 0; nt < NT; ++nt) b1[nt] = load_b_frag16(bptr[nt] + k0 + 32);
        // WMMA on stage 0
#pragma unroll
        for (int nt = 0; nt < NT; ++nt)
#pragma unroll
            for (int mt = 0; mt < MT; ++mt)
                acc[mt][nt] = __builtin_amdgcn_wmma_f32_16x16x32_f16(
                    false, a0[mt], false, b0[nt], (short)0, acc[mt][nt], false, false);
        // stage 0 loads for next iteration (k0+64)
        if (k0 + 64 < K) {
#pragma unroll
            for (int mt = 0; mt < MT; ++mt) a0[mt] = load_a_frag16(aptr[mt] + k0 + 64);
#pragma unroll
            for (int nt = 0; nt < NT; ++nt) b0[nt] = load_b_frag16(bptr[nt] + k0 + 64);
        }
        // WMMA on stage 1
#pragma unroll
        for (int nt = 0; nt < NT; ++nt)
#pragma unroll
            for (int mt = 0; mt < MT; ++mt)
                acc[mt][nt] = __builtin_amdgcn_wmma_f32_16x16x32_f16(
                    false, a1[mt], false, b1[nt], (short)0, acc[mt][nt], false, false);
    }

    // D element (r, lane): M = mbase + r, N = n0 + nt*16 + (lane&15)
#pragma unroll
    for (int mt = 0; mt < MT; ++mt) {
        const int mbase = m0 + mt * 16 + 8 * (lane >> 4);
#pragma unroll
        for (int nt = 0; nt < NT; ++nt) {
            const int n = n0 + nt * 16 + (lane & 15);
#pragma unroll
            for (int r = 0; r < 8; ++r) {
                const int m = mbase + r;
                float v = acc[mt][nt][r];
                if (EPI == 0) {
                    Dst[(size_t)m * ldd + n] = v;
                } else if (EPI == 1) {
                    Dst[(size_t)n * ldd + m] = v;
                } else {
                    float x  = v + bias[n];
                    float sp = (x > 20.f) ? x : log1pf(__expf(x));
                    Dst[(size_t)n * ldd + m] = sp;
                }
            }
        }
    }
}

// bulk f32 -> f16 (4 elements per thread, vectorized)
__global__ void f32_to_f16_kernel(const float* __restrict__ s,
                                  _Float16* __restrict__ d, int n4)
{
    int i = blockIdx.x * blockDim.x + threadIdx.x;
    if (i >= n4) return;
    float4 v = ((const float4*)s)[i];
    v4h h;
    h[0] = (_Float16)v.x; h[1] = (_Float16)v.y;
    h[2] = (_Float16)v.z; h[3] = (_Float16)v.w;
    ((v4h*)d)[i] = h;
}

// causal depthwise conv1d (k=4, left pad 3) + SiLU
// -> convT (f32, (d,l)) for the scan, convLDh (f16, (l,d)) for x_proj GEMM
__global__ void conv_silu_kernel(const float* __restrict__ projT,
                                 const float* __restrict__ cw,
                                 const float* __restrict__ cb,
                                 float* __restrict__ convT,
                                 _Float16* __restrict__ convLDh)
{
    int idx = blockIdx.x * blockDim.x + threadIdx.x;
    if (idx >= DI * LSEQ) return;
    int l = idx & (LSEQ - 1);
    int d = idx >> 11;
    const float* h = projT + (size_t)d * LSEQ;
    float acc = cb[d];
#pragma unroll
    for (int j = 0; j < 4; ++j) {
        int ls = l + j - 3;
        float hv = (ls >= 0) ? h[ls] : 0.f;
        acc = fmaf(cw[d * 4 + j], hv, acc);
    }
    float s = acc / (1.f + __expf(-acc));   // SiLU
    convT[idx] = s;
    convLDh[(size_t)l * DI + d] = (_Float16)s;
}

// zero-pad dt_proj operands to K=64 (f16): Apadh (2048,64) from ssm[:, :48],
// Bpadh (1536,64) from dt_w (1536,48)
__global__ void pad_dt_kernel(const float* __restrict__ ssm,
                              const float* __restrict__ dt_w,
                              _Float16* __restrict__ Apadh,
                              _Float16* __restrict__ Bpadh)
{
    int idx = blockIdx.x * blockDim.x + threadIdx.x;
    if (idx < LSEQ * KDT) {
        int col = idx & (KDT - 1), row = idx >> 6;
        Apadh[idx] = (col < DTR) ? (_Float16)ssm[(size_t)row * SSMW + col] : (_Float16)0.f;
    }
    if (idx < DI * KDT) {
        int col = idx & (KDT - 1), row = idx >> 6;
        Bpadh[idx] = (col < DTR) ? (_Float16)dt_w[(size_t)row * DTR + col] : (_Float16)0.f;
    }
}

// Bidirectional selective scan: 16 lanes per d-channel (one per state n).
__global__ __launch_bounds__(128)
void scan_kernel(const float* __restrict__ dtT,
                 const float* __restrict__ convT,
                 const float* __restrict__ ssm,
                 const float* __restrict__ A_log,
                 const float* __restrict__ Dvec,
                 const float* __restrict__ gateT,
                 float* __restrict__ scanA,
                 _Float16* __restrict__ yLDh)
{
    const int tid = threadIdx.x;
    const int d   = blockIdx.x * 8 + (tid >> 4);
    const int n   = tid & 15;
    const float Aneg = -__expf(A_log[d * NST + n]);
    const float Dval = Dvec[d];
    const float* dtd = dtT   + (size_t)d * LSEQ;
    const float* hxd = convT + (size_t)d * LSEQ;

    // forward: h[l] = dA*h[l-1] + dt*B*u ; scanA = h.C + D*hidden
    float state = 0.f;
    for (int l = 0; l < LSEQ; ++l) {
        float sdt = dtd[l];
        float hx  = hxd[l];
        float Bv  = ssm[(size_t)l * SSMW + DTR + n];
        float Cv  = ssm[(size_t)l * SSMW + DTR + NST + n];
        float dA  = __expf(Aneg * sdt);
        state = fmaf(dA, state, sdt * Bv * hx);
        float contrib = state * Cv;
        contrib += __shfl_xor(contrib, 8, 16);
        contrib += __shfl_xor(contrib, 4, 16);
        contrib += __shfl_xor(contrib, 2, 16);
        contrib += __shfl_xor(contrib, 1, 16);
        if (n == 0) scanA[(size_t)d * LSEQ + l] = contrib + Dval * hx;
    }

    // backward: g[l] = dA[l+1]*(x[l+1]+g[l+1]) ; finalize + SiLU gate -> yLDh
    float g = 0.f;
    for (int l = LSEQ - 1; l >= 0; --l) {
        if (l < LSEQ - 1) {
            float sdt1 = dtd[l + 1];
            float hx1  = hxd[l + 1];
            float Bv1  = ssm[(size_t)(l + 1) * SSMW + DTR + n];
            float dA1  = __expf(Aneg * sdt1);
            g = dA1 * (g + sdt1 * Bv1 * hx1);
        }
        float Cv = ssm[(size_t)l * SSMW + DTR + NST + n];
        float contrib = g * Cv;
        contrib += __shfl_xor(contrib, 8, 16);
        contrib += __shfl_xor(contrib, 4, 16);
        contrib += __shfl_xor(contrib, 2, 16);
        contrib += __shfl_xor(contrib, 1, 16);
        if (n == 0) {
            float tot = scanA[(size_t)d * LSEQ + l] + contrib;
            float gt  = gateT[(size_t)d * LSEQ + l];
            float y   = tot * (gt / (1.f + __expf(-gt)));   // * silu(gate)
            yLDh[(size_t)l * DI + d] = (_Float16)y;
        }
    }
}

extern "C" void kernel_launch(void* const* d_in, const int* in_sizes, int n_in,
                              void* d_out, int out_size, void* d_ws, size_t ws_size,
                              hipStream_t stream)
{
    const float* x       = (const float*)d_in[0];  // (2048, 768)
    const float* in_w    = (const float*)d_in[1];  // (3072, 768)
    const float* conv_w  = (const float*)d_in[2];  // (1536, 1, 4)
    const float* conv_b  = (const float*)d_in[3];  // (1536,)
    const float* xproj_w = (const float*)d_in[4];  // (80, 1536)
    const float* dt_w    = (const float*)d_in[5];  // (1536, 48)
    const float* dt_b    = (const float*)d_in[6];  // (1536,)
    const float* A_log   = (const float*)d_in[7];  // (1536, 16)
    const float* Dvec    = (const float*)d_in[8];  // (1536,)
    const float* out_w   = (const float*)d_in[9];  // (768, 1536)
    float*       out     = (float*)d_out;          // (2048, 768)

    // ---- workspace carve-up (f32 region, then f16 region; all 32B aligned) --
    float* ws     = (float*)d_ws;
    float* projT  = ws;                                 // 3072*2048 f32
    float* convT  = projT  + (size_t)2 * DI * LSEQ;     // 1536*2048 f32
    float* ssm    = convT  + (size_t)DI * LSEQ;         // 2048*80   f32
    float* dtT    = ssm    + (size_t)LSEQ * SSMW;       // 1536*2048 f32
    float* scanA  = dtT    + (size_t)DI * LSEQ;         // 1536*2048 f32
    float* gateT  = projT  + (size_t)DI * LSEQ;         // rows 1536.. of projT

    _Float16* xh      = (_Float16*)(scanA + (size_t)DI * LSEQ);
    _Float16* inwh    = xh      + (size_t)LSEQ * DM;    // 3072*768
    _Float16* xwh     = inwh    + (size_t)2 * DI * DM;  // 80*1536
    _Float16* owh     = xwh     + (size_t)SSMW * DI;    // 768*1536
    _Float16* convLDh = owh     + (size_t)DM * DI;      // 2048*1536
    _Float16* Apadh   = convLDh + (size_t)LSEQ * DI;    // 2048*64
    _Float16* Bpadh   = Apadh   + (size_t)LSEQ * KDT;   // 1536*64
    _Float16* yLDh    = Bpadh   + (size_t)DI * KDT;     // 2048*1536

    // 0) one-time f32 -> f16 conversions of GEMM operands
    f32_to_f16_kernel<<<(LSEQ * DM / 4 + 255) / 256, 256, 0, stream>>>(x, xh, LSEQ * DM / 4);
    f32_to_f16_kernel<<<(2 * DI * DM / 4 + 255) / 256, 256, 0, stream>>>(in_w, inwh, 2 * DI * DM / 4);
    f32_to_f16_kernel<<<(SSMW * DI / 4 + 255) / 256, 256, 0, stream>>>(xproj_w, xwh, SSMW * DI / 4);
    f32_to_f16_kernel<<<(DM * DI / 4 + 255) / 256, 256, 0, stream>>>(out_w, owh, DM * DI / 4);

    // 1) in_proj -> projT (transposed, f32)
    wmma_gemm<2, 4, DM, 1><<<dim3((2 * DI) / 64, LSEQ / 32), 32, 0, stream>>>(
        xh, DM, inwh, DM, nullptr, projT, LSEQ);

    // 2) depthwise conv + SiLU
    conv_silu_kernel<<<(DI * LSEQ + 255) / 256, 256, 0, stream>>>(
        projT, conv_w, conv_b, convT, convLDh);

    // 3) x_proj -> ssm (2048,80) row-major f32
    wmma_gemm<2, 5, DI, 0><<<dim3(1, LSEQ / 32), 32, 0, stream>>>(
        convLDh, DI, xwh, DI, nullptr, ssm, SSMW);

    // 4a) zero-pad dt_proj operands (f16, K=64)
    pad_dt_kernel<<<(LSEQ * KDT + 255) / 256, 256, 0, stream>>>(
        ssm, dt_w, Apadh, Bpadh);

    // 4b) dt_proj + bias + softplus -> dtT (transposed, f32)
    wmma_gemm<2, 4, KDT, 2><<<dim3(DI / 64, LSEQ / 32), 32, 0, stream>>>(
        Apadh, KDT, Bpadh, KDT, dt_b, dtT, LSEQ);

    // 5) bidirectional scan + D*hidden + gate -> yLDh (f16)
    scan_kernel<<<DI / 8, 128, 0, stream>>>(
        dtT, convT, ssm, A_log, Dvec, gateT, scanA, yLDh);

    // 6) out_proj -> out (2048,768) f32
    wmma_gemm<2, 4, DI, 0><<<dim3(DM / 64, LSEQ / 32), 32, 0, stream>>>(
        yLDh, DI, owh, DI, nullptr, out, DM);
}
